// SparseCoder_83580063580811
// MI455X (gfx1250) — matrix operand order repaired and checked by reading, not compiled, not run on hardware
//
#include <hip/hip_runtime.h>
#include <hip/hip_bf16.h>

// ---------------------------------------------------------------------------
// TopK Sparse Autoencoder forward for MI455X (gfx1250, wave32, WMMA).
// Encoder GEMM (0.55 TFLOP) in bf16 WMMA w/ f32 accum; W_enc bf16 (128 MB)
// is L2-resident (192 MB L2). GEMM uses a 2x2 WMMA register tile per wave
// (32 rows x 32 latents) -> 2 b128 loads per WMMA, 4 independent acc chains.
// All reductions are fixed-order trees (deterministic across replays).
// ---------------------------------------------------------------------------

typedef __attribute__((ext_vector_type(16))) __bf16 v16bf;
typedef __attribute__((ext_vector_type(8)))  float  v8f;
typedef __attribute__((ext_vector_type(4)))  int    v4i;
typedef __attribute__((ext_vector_type(8)))  int    v8i;

struct APair { v4i lo, hi; };   // 32B: the two 16B halves of an A fragment

__device__ __forceinline__ unsigned short f2bf(float f) {
    // round-to-nearest-even fp32 -> bf16
    unsigned int u = __float_as_uint(f);
    u += 0x7FFFu + ((u >> 16) & 1u);
    return (unsigned short)(u >> 16);
}

__device__ __forceinline__ v16bf load_afrag(const unsigned short* p) {
    APair ap;
    ap.lo = *(const v4i*)(p);        // K {0..7}  (lanes>=16: {16..23})
    ap.hi = *(const v4i*)(p + 8);    // K {8..15} (lanes>=16: {24..31})
    return __builtin_bit_cast(v16bf, ap);
}

// ---- (1a) x - b_dec  ->  bf16 -------------------------------------------------
__global__ __launch_bounds__(256) void cvt_x_kernel(
    const float* __restrict__ x, const float* __restrict__ b_dec,
    unsigned short* __restrict__ xbf, int D, size_t total)
{
    size_t i = (size_t)blockIdx.x * 256 + threadIdx.x;
    if (i < total) {
        int d = (int)(i % (size_t)D);
        xbf[i] = f2bf(x[i] - b_dec[d]);
    }
}

// ---- (1b) W_enc -> bf16 -------------------------------------------------------
__global__ __launch_bounds__(256) void cvt_w_kernel(
    const float* __restrict__ w, unsigned short* __restrict__ wbf, size_t total)
{
    size_t i = (size_t)blockIdx.x * 256 + threadIdx.x;
    if (i < total) wbf[i] = f2bf(w[i]);
}

// ---- (2) Encoder GEMM: pre = relu(Xbf @ Wbf^T + b_enc) ------------------------
// Block = 256 threads = 8 waves. Block tile: 32 rows x 256 latents.
// Each wave: 2x2 WMMA register tile (32 rows x 32 latents), 4 independent
// accumulator chains; A fragments shared across column tiles and vice versa
// -> 8 b128 loads per 4 WMMAs.
__global__ __launch_bounds__(256) void enc_gemm_kernel(
    const unsigned short* __restrict__ Xbf,   // [N, D] bf16
    const unsigned short* __restrict__ Wbf,   // [L, D] bf16
    const float* __restrict__ b_enc,          // [L]
    float* __restrict__ pre,                  // [N, L]
    int D, int L)
{
    const int lane = threadIdx.x & 31;
    const int wave = threadIdx.x >> 5;
    const int n0   = blockIdx.y << 5;                       // 32 rows
    const int l0   = (blockIdx.x << 8) + (wave << 5);       // 32 latents / wave
    const int r    = lane & 15;
    const int h    = lane >> 4;

    // A fragments: row = r (+16 for tile 1); lanes<16 -> K {0..15},
    // lanes>=16 -> K {16..31}  (contiguous, matches B)
    const unsigned short* aptr0 = Xbf + (size_t)(n0 + r) * D + h * 16;
    const unsigned short* aptr1 = aptr0 + (size_t)16 * D;
    // B fragments: col = r (a W_enc row, contiguous over D); lanes>=16 -> K+16
    const unsigned short* bptr0 = Wbf + (size_t)(l0 + r) * D + h * 16;
    const unsigned short* bptr1 = bptr0 + (size_t)16 * D;

    v8f acc00 = {}; v8f acc01 = {};   // row-tile 0 x col-tiles {0,1}
    v8f acc10 = {}; v8f acc11 = {};   // row-tile 1 x col-tiles {0,1}
    for (int k = 0; k < D; k += 32) {
        __builtin_prefetch(bptr0 + k + 512, 0, 1);   // global_prefetch_b8, B stream
        __builtin_prefetch(bptr1 + k + 512, 0, 1);
        v16bf a0 = load_afrag(aptr0 + k);
        v16bf a1 = load_afrag(aptr1 + k);
        v16bf b0 = __builtin_bit_cast(v16bf, *(const v8i*)(bptr0 + k));
        v16bf b1 = __builtin_bit_cast(v16bf, *(const v8i*)(bptr1 + k));
        acc00 = __builtin_amdgcn_wmma_f32_16x16x32_bf16(
                    false, a0, false, b0, (short)0, acc00, false, false);
        acc01 = __builtin_amdgcn_wmma_f32_16x16x32_bf16(
                    false, a0, false, b1, (short)0, acc01, false, false);
        acc10 = __builtin_amdgcn_wmma_f32_16x16x32_bf16(
                    false, a1, false, b0, (short)0, acc10, false, false);
        acc11 = __builtin_amdgcn_wmma_f32_16x16x32_bf16(
                    false, a1, false, b1, (short)0, acc11, false, false);
    }

    // Epilogue: bias + ReLU. C layout: VGPR i -> row (i + h*8), col = r (per lane)
    const int col0 = l0 + r, col1 = col0 + 16;
    const float bias0 = b_enc[col0], bias1 = b_enc[col1];
#pragma unroll
    for (int i = 0; i < 8; ++i) {
        int row0 = n0 + i + h * 8;
        int row1 = row0 + 16;
        float v00 = acc00[i] + bias0; v00 = v00 > 0.f ? v00 : 0.f;
        float v01 = acc01[i] + bias1; v01 = v01 > 0.f ? v01 : 0.f;
        float v10 = acc10[i] + bias0; v10 = v10 > 0.f ? v10 : 0.f;
        float v11 = acc11[i] + bias1; v11 = v11 > 0.f ? v11 : 0.f;
        pre[(size_t)row0 * L + col0] = v00;
        pre[(size_t)row0 * L + col1] = v01;
        pre[(size_t)row1 * L + col0] = v10;
        pre[(size_t)row1 * L + col1] = v11;
    }
}

// ---- (3) Per-row top-32: row cached in LDS (L*4 = 128 KB of 320 KB WGP LDS) ---
// 32 argmax passes; tie-break = lowest index (matches jax.lax.top_k).
__global__ __launch_bounds__(256) void topk_kernel(
    const float* __restrict__ pre, float* __restrict__ top_acts,
    int* __restrict__ top_idx, int L)
{
    extern __shared__ float row[];
    __shared__ float rv[256];
    __shared__ int   ri[256];
    const int n = blockIdx.x, tid = threadIdx.x;
    const float* src = pre + (size_t)n * L;
    for (int j = tid; j < L; j += 256) row[j] = src[j];
    __syncthreads();

    for (int it = 0; it < 32; ++it) {
        float bv = -1.0f; int bi = 0x7fffffff;
        for (int j = tid; j < L; j += 256) {
            float v = row[j];
            if (v > bv) { bv = v; bi = j; }      // strict > keeps lowest j in-thread
        }
        rv[tid] = bv; ri[tid] = bi;
        __syncthreads();
        for (int s = 128; s > 0; s >>= 1) {
            if (tid < s) {
                float ov = rv[tid + s]; int oi = ri[tid + s];
                if (ov > rv[tid] || (ov == rv[tid] && oi < ri[tid])) {
                    rv[tid] = ov; ri[tid] = oi;
                }
            }
            __syncthreads();
        }
        if (tid == 0) {
            top_acts[(size_t)n * 32 + it] = rv[0];
            top_idx [(size_t)n * 32 + it] = ri[0];
            row[ri[0]] = -1.0f;                  // knock out the winner
        }
        __syncthreads();
    }
}

// ---- (4) Decoder gather + per-row error partial -------------------------------
__global__ __launch_bounds__(256) void decode_kernel(
    const float* __restrict__ x, const float* __restrict__ Wd,
    const float* __restrict__ b_dec, const float* __restrict__ top_acts,
    const int* __restrict__ top_idx, float* __restrict__ out,
    float* __restrict__ err_part, int D)
{
    __shared__ float sw[32];
    __shared__ int   si[32];
    __shared__ float red[256];
    const int n = blockIdx.x, tid = threadIdx.x;
    if (tid < 32) {
        sw[tid] = top_acts[(size_t)n * 32 + tid];
        si[tid] = top_idx [(size_t)n * 32 + tid];
    }
    __syncthreads();

    const int ncol = D >> 8;                     // D/256 columns per thread (<=8)
    float acc[8];
    for (int j = 0; j < ncol; ++j) acc[j] = 0.f;
    for (int kk = 0; kk < 32; ++kk) {
        const float w = sw[kk];
        const float* wr = Wd + (size_t)si[kk] * D;
        for (int j = 0; j < ncol; ++j) acc[j] += w * wr[tid + (j << 8)];
    }
    float esum = 0.f;
    for (int j = 0; j < ncol; ++j) {
        int d = tid + (j << 8);
        float o = acc[j] + b_dec[d];
        out[(size_t)n * D + d] = o;
        float e = x[(size_t)n * D + d] - o;
        esum += e * e;
    }
    red[tid] = esum; __syncthreads();
    for (int s = 128; s > 0; s >>= 1) {
        if (tid < s) red[tid] += red[tid + s];
        __syncthreads();
    }
    if (tid == 0) err_part[n] = red[0];
}

// ---- (5) Column mean + variance partials (one column per thread, two passes) --
__global__ __launch_bounds__(256) void colvar_kernel(
    const float* __restrict__ x, float* __restrict__ var_part, int N, int D)
{
    __shared__ float red[256];
    const int d = blockIdx.x * 256 + threadIdx.x;
    float s = 0.f;
    for (int nn = 0; nn < N; ++nn) s += x[(size_t)nn * D + d];
    const float m = s / (float)N;
    float v = 0.f;
    for (int nn = 0; nn < N; ++nn) {
        float t = x[(size_t)nn * D + d] - m;
        v += t * t;
    }
    red[threadIdx.x] = v; __syncthreads();
    for (int s2 = 128; s2 > 0; s2 >>= 1) {
        if (threadIdx.x < s2) red[threadIdx.x] += red[threadIdx.x + s2];
        __syncthreads();
    }
    if (threadIdx.x == 0) var_part[blockIdx.x] = red[0];
}

// ---- (6) Final scalar: fvu = sum(e^2) / total_variance ; auxk = multi = 0 -----
__global__ __launch_bounds__(256) void finalize_kernel(
    const float* __restrict__ err_part, int nerr,
    const float* __restrict__ var_part, int nvar,
    float* __restrict__ out_scalars)
{
    __shared__ float red[256];
    const int tid = threadIdx.x;
    float s = 0.f;
    for (int i = tid; i < nerr; i += 256) s += err_part[i];
    red[tid] = s; __syncthreads();
    for (int s2 = 128; s2 > 0; s2 >>= 1) {
        if (tid < s2) red[tid] += red[tid + s2];
        __syncthreads();
    }
    const float errsum = red[0];
    __syncthreads();
    float v = 0.f;
    for (int i = tid; i < nvar; i += 256) v += var_part[i];
    red[tid] = v; __syncthreads();
    for (int s2 = 128; s2 > 0; s2 >>= 1) {
        if (tid < s2) red[tid] += red[tid + s2];
        __syncthreads();
    }
    if (tid == 0) {
        out_scalars[0] = errsum / red[0];   // fvu
        out_scalars[1] = 0.f;               // auxk_loss
        out_scalars[2] = 0.f;               // multi_topk_fvu
    }
}

static inline size_t align256(size_t v) { return (v + 255) & ~(size_t)255; }

extern "C" void kernel_launch(void* const* d_in, const int* in_sizes, int n_in,
                              void* d_out, int out_size, void* d_ws, size_t ws_size,
                              hipStream_t stream)
{
    (void)n_in; (void)out_size; (void)ws_size;
    const float* x     = (const float*)d_in[0];
    const float* W_enc = (const float*)d_in[1];
    const float* b_enc = (const float*)d_in[2];
    const float* W_dec = (const float*)d_in[3];
    const float* b_dec = (const float*)d_in[4];

    const int D = in_sizes[4];            // b_dec length  (2048)
    const int L = in_sizes[2];            // b_enc length  (32768)
    const int N = in_sizes[0] / D;        // rows          (4096)
    const int K = 32;                     // reference k

    // Workspace layout
    char* ws = (char*)d_ws;
    size_t off = 0;
    unsigned short* xbf = (unsigned short*)(ws + off); off = align256(off + (size_t)N * D * 2);
    unsigned short* wbf = (unsigned short*)(ws + off); off = align256(off + (size_t)L * D * 2);
    float* pre          = (float*)(ws + off);          off = align256(off + (size_t)N * L * 4);
    float* err_part     = (float*)(ws + off);          off = align256(off + (size_t)N * 4);
    float* var_part     = (float*)(ws + off);          off = align256(off + (size_t)(D / 256) * 4);

    // Output layout (concatenated flat, reference return order)
    float* out      = (float*)d_out;
    float* sae_out  = out;
    float* top_acts = out + (size_t)N * D;
    int*   top_idx  = (int*)(out + (size_t)N * D + (size_t)N * K);
    float* scalars  = out + (size_t)N * D + 2 * (size_t)N * K;

    const size_t xtot = (size_t)N * D;
    const size_t wtot = (size_t)L * D;
    cvt_x_kernel<<<(unsigned)((xtot + 255) / 256), 256, 0, stream>>>(x, b_dec, xbf, D, xtot);
    cvt_w_kernel<<<(unsigned)((wtot + 255) / 256), 256, 0, stream>>>(W_enc, wbf, wtot);

    dim3 gemm_grid((unsigned)(L / 256), (unsigned)(N / 32));
    enc_gemm_kernel<<<gemm_grid, 256, 0, stream>>>(xbf, wbf, b_enc, pre, D, L);

    topk_kernel<<<N, 256, (size_t)L * sizeof(float), stream>>>(pre, top_acts, top_idx, L);

    decode_kernel<<<N, 256, 0, stream>>>(x, W_dec, b_dec, top_acts, top_idx,
                                         sae_out, err_part, D);

    colvar_kernel<<<D / 256, 256, 0, stream>>>(x, var_part, N, D);

    finalize_kernel<<<1, 256, 0, stream>>>(err_part, N, var_part, D / 256, scalars);
}
